// HoloNetVault_11519102288304
// MI455X (gfx1250) — compile-verified, matrix operand order checked by reading, not run on hardware
//
#include <hip/hip_runtime.h>
#include <hip/hip_bf16.h>
#include <math.h>

// ---------------------------------------------------------------------------
// HoloNetVault recurrence for MI455X (gfx1250, wave32, WMMA)
//   D = (I+S)^-1 (I-S) via Newton-Schulz (fp32 WMMA 16x16x4, pure load+wmma
//   inner loop; residual 2I-T hoisted to an elementwise pass)
//   g,u projections via bf16 WMMA 16x16x32 (weights pre-converted to bf16)
//   sequential scan: D split column-wise over 32 WGPs, D-slices resident in
//   LDS, per-step cross-WGP sync via sense-reversing atomic barrier.
// ---------------------------------------------------------------------------

typedef __attribute__((ext_vector_type(2)))  float  v2f;
typedef __attribute__((ext_vector_type(8)))  float  v8f;
typedef __attribute__((ext_vector_type(8)))  __bf16 v8bf;
typedef __attribute__((ext_vector_type(16))) __bf16 v16bf;

#define D_MODEL 1024
#define SEQ     4096
#define BATCH   8
#define RANK    16
#define N_WG    32          // scan workgroups (1 WGP each, big LDS)
#define COLS_WG 32          // D columns per scan workgroup
#define DT_STR  1028        // LDS row stride (4*r+delta covers all 64 banks)
#define NEWTON_ITERS 6

__device__ __forceinline__ v8f wmma_f32x4(v2f a, v2f b, v8f c) {
  // D = A(16x4,f32) * B(4x16,f32) + C(16x16,f32)
  return __builtin_amdgcn_wmma_f32_16x16x4_f32(false, a, false, b,
                                               (short)0, c, false, false);
}

__device__ __forceinline__ v8f wmma_bf16(v16bf a, v16bf b, v8f c) {
  // D = A(16x32,bf16) * B(32x16,bf16) + C(16x16,f32)
  return __builtin_amdgcn_wmma_f32_16x16x32_bf16(false, a, false, b,
                                                 (short)0, c, false, false);
}

__device__ __forceinline__ float sigmoidf_(float z) {
  return 1.0f / (1.0f + expf(-z));
}

// bf16 fragment slice from an f32 row (on-the-fly convert, used for x only).
// elements 0..7 <- k = kbase..kbase+7 ; elements 8..15 <- k = kbase+16..+23
__device__ __forceinline__ v16bf frag_cvt(const float* __restrict__ row, int kbase) {
  v16bf r;
#pragma unroll
  for (int e = 0; e < 8; ++e) r[e]     = (__bf16)row[kbase + e];
#pragma unroll
  for (int e = 0; e < 8; ++e) r[8 + e] = (__bf16)row[kbase + 16 + e];
  return r;
}

// bf16 fragment slice straight from a pre-converted bf16 row: two b128 loads.
__device__ __forceinline__ v16bf frag_ld(const __bf16* __restrict__ row, int kbase) {
  const v8bf lo = *(const v8bf*)(row + kbase);
  const v8bf hi = *(const v8bf*)(row + kbase + 16);
  return __builtin_shufflevector(lo, hi, 0, 1, 2, 3, 4, 5, 6, 7,
                                 8, 9, 10, 11, 12, 13, 14, 15);
}

// ---------------------------------------------------------------------------
// Setup: S = 0.5*(Sp - Sp^T); M = I+S; N0 = I-S; X1 = 2I - M = I - S (first
// Newton step from X0 = I). Also zero h-state (16x1024, rows 8..15 stay 0
// forever) and reset the grid-barrier counters (graph-replay safe).
// ---------------------------------------------------------------------------
__global__ void k_setup(const float* __restrict__ Sp,
                        float* __restrict__ Mb, float* __restrict__ N0,
                        float* __restrict__ X1, float* __restrict__ HG,
                        unsigned* __restrict__ bar) {
  const int gid = blockIdx.x * blockDim.x + threadIdx.x;   // 1M threads
  const int i = gid >> 10, j = gid & 1023;
  const float s = 0.5f * (Sp[i * D_MODEL + j] - Sp[j * D_MODEL + i]);
  const float eye = (i == j) ? 1.0f : 0.0f;
  Mb[gid] = eye + s;
  const float n0 = eye - s;
  N0[gid] = n0;
  X1[gid] = n0;                     // X after one Newton step
  if (gid < 16 * D_MODEL) HG[gid] = 0.0f;
  if (gid == 0) { bar[0] = 0u; bar[1] = 0u; }
}

// In-place Newton residual: T <- 2I - T   (keeps GEMM inner loop pure)
__global__ void k_neg2i(float* __restrict__ T) {
  const int gid = blockIdx.x * blockDim.x + threadIdx.x;   // 1M threads
  const float eye2 = ((gid >> 10) == (gid & 1023)) ? 2.0f : 0.0f;
  T[gid] = eye2 - T[gid];
}

// One-time f32 -> bf16 conversion of the two projection weight matrices.
__global__ void k_cvt_w(const float* __restrict__ gw, const float* __restrict__ ww,
                        __bf16* __restrict__ gwb, __bf16* __restrict__ wwb) {
  const int gid = blockIdx.x * blockDim.x + threadIdx.x;   // 1M threads
  gwb[gid] = (__bf16)gw[gid];
  wwb[gid] = (__bf16)ww[gid];
}

// ---------------------------------------------------------------------------
// fp32 GEMM (1024x1024x1024, row-major): C = A * B.
// Each wave: 16(M) x 64(N) strip, K streamed 4 at a time through wmma.
// 128 blocks x 256 threads = 1024 waves = 64 M-strips x 16 N-strips.
// ---------------------------------------------------------------------------
__global__ void k_gemm_nn(float* __restrict__ C, const float* __restrict__ A,
                          const float* __restrict__ B) {
  const int lane = threadIdx.x & 31;
  const int wid  = blockIdx.x * 8 + (threadIdx.x >> 5);
  const int m0 = (wid >> 4) << 4;
  const int n0 = (wid & 15) << 6;
  const int l15 = lane & 15;
  const int kh2 = (lane >> 4) << 1;          // K offset 0 or 2 within step
  const float* arow = A + (size_t)(m0 + l15) * D_MODEL;
  v8f acc[4] = {};
  for (int k0 = 0; k0 < D_MODEL; k0 += 4) {
    const int kk = k0 + kh2;
    v2f av; av.x = arow[kk]; av.y = arow[kk + 1];
#pragma unroll
    for (int nt = 0; nt < 4; ++nt) {
      const int col = n0 + (nt << 4) + l15;
      v2f bv;
      bv.x = B[(size_t)kk * D_MODEL + col];
      bv.y = B[(size_t)(kk + 1) * D_MODEL + col];
      acc[nt] = wmma_f32x4(av, bv, acc[nt]);
    }
  }
  const int mrow = m0 + ((lane >> 4) << 3);  // + vgpr index i below
#pragma unroll
  for (int nt = 0; nt < 4; ++nt) {
    const int col = n0 + (nt << 4) + l15;
#pragma unroll
    for (int i = 0; i < 8; ++i)
      C[(size_t)(mrow + i) * D_MODEL + col] = acc[nt][i];
  }
}

// ---------------------------------------------------------------------------
// Projections: g = sigmoid(x @ gw^T + gb) -> G ;  u = x @ ww^T + wb -> U
// bf16 WMMA 16x16x32. Weights pre-converted to bf16 (pure b128 loads in the
// hot loop); x converted on the fly. Wave strip = 16(M) x 64(N), x fragment
// shared by both weight matrices. 4096 blocks x 256 threads.
// ---------------------------------------------------------------------------
__global__ void k_proj(const float* __restrict__ x,
                       const __bf16* __restrict__ gwb, const float* __restrict__ gb,
                       const __bf16* __restrict__ wwb, const float* __restrict__ wb,
                       float* __restrict__ G, float* __restrict__ U) {
  const int lane = threadIdx.x & 31;
  const int w = threadIdx.x >> 5;
  const int m0 = (blockIdx.x >> 4) * 128 + w * 16;
  const int n0 = (blockIdx.x & 15) * 64;
  const int l15 = lane & 15;
  const int kofs = (lane >> 4) << 3;         // 0 or 8
  const float* xrow = x + (size_t)(m0 + l15) * D_MODEL;
  v8f ga[4] = {}, ua[4] = {};
  for (int kc = 0; kc < D_MODEL; kc += 32) {
    const v16bf af = frag_cvt(xrow, kc + kofs);
#pragma unroll
    for (int nt = 0; nt < 4; ++nt) {
      const int col = n0 + (nt << 4) + l15;
      const v16bf bg = frag_ld(gwb + (size_t)col * D_MODEL, kc + kofs);
      const v16bf bw = frag_ld(wwb + (size_t)col * D_MODEL, kc + kofs);
      ga[nt] = wmma_bf16(af, bg, ga[nt]);
      ua[nt] = wmma_bf16(af, bw, ua[nt]);
    }
  }
  const int mrow = m0 + ((lane >> 4) << 3);
#pragma unroll
  for (int nt = 0; nt < 4; ++nt) {
    const int col = n0 + (nt << 4) + l15;
    const float gbv = gb[col], wbv = wb[col];
#pragma unroll
    for (int i = 0; i < 8; ++i) {
      const size_t idx = (size_t)(mrow + i) * D_MODEL + col;
      G[idx] = sigmoidf_(ga[nt][i] + gbv);
      U[idx] = ua[nt][i] + wbv;
    }
  }
}

// ---------------------------------------------------------------------------
// Sequential scan. 32 WGs x 256 threads; WG b owns columns [32b, 32b+32).
// LDS: D^T slice (32x1024, stride 1028), staged h (16x1024, stride 1028),
// 12 partial tiles, hB tile.  Per step:
//   hB  = h @ B^T           (8 waves split K, LDS reduce)
//   acc = h @ D^T[:,cols]   (2 N-tiles x 4 K-quarters, LDS reduce)
//   lr  = hB @ A^T[:,cols]  (waves 0-1)
//   h'  = gd*acc + g_t*lr + u_t   -> ys[t] (in-place over u) and h-state
// Cross-WGP sync: sense-reversing atomic barrier (+HW cluster barrier, which
// is an architectural NOP when not dispatched as a cluster).
// ---------------------------------------------------------------------------
__device__ __forceinline__ void grid_barrier(unsigned* bar, int nwg) {
  __syncthreads();
  __threadfence();
  if (threadIdx.x == 0) {
    unsigned* ctr = bar;
    unsigned* gen = bar + 1;
    const unsigned g = __hip_atomic_load(gen, __ATOMIC_RELAXED,
                                         __HIP_MEMORY_SCOPE_AGENT);
    const unsigned old = __hip_atomic_fetch_add(ctr, 1u, __ATOMIC_ACQ_REL,
                                                __HIP_MEMORY_SCOPE_AGENT);
    if (old == (unsigned)(nwg - 1)) {
      __hip_atomic_store(ctr, 0u, __ATOMIC_RELAXED, __HIP_MEMORY_SCOPE_AGENT);
      __hip_atomic_fetch_add(gen, 1u, __ATOMIC_RELEASE,
                             __HIP_MEMORY_SCOPE_AGENT);
    } else {
      while (__hip_atomic_load(gen, __ATOMIC_ACQUIRE,
                               __HIP_MEMORY_SCOPE_AGENT) == g)
        __builtin_amdgcn_s_sleep(1);
    }
  }
  __threadfence();
  __syncthreads();
  __builtin_amdgcn_s_cluster_barrier();   // NOP outside a cluster dispatch
}

__global__ void k_scan(const float* __restrict__ Dm, const float* __restrict__ Amat,
                       const float* __restrict__ Bmat, const float* __restrict__ G,
                       float* __restrict__ Y, float* __restrict__ HG,
                       const float* __restrict__ gamma, unsigned* __restrict__ bar) {
  extern __shared__ float sm[];
  float* DT = sm;                              // 32 * 1028
  float* HL = DT + COLS_WG * DT_STR;           // 16 * 1028
  float* SC = HL + 16 * DT_STR;                // 12 * 256 partial tiles
  float* HB = SC + 12 * 256;                   // 16 * 20
  const int tid  = threadIdx.x;
  const int w    = tid >> 5;
  const int lane = tid & 31;
  const int l15  = lane & 15;
  const int kh2  = (lane >> 4) << 1;
  const int c0   = blockIdx.x * COLS_WG;
  const float gd = sigmoidf_(gamma[0]);

  // Cache this WG's 32 rows of Dmat (== columns of D^T) in LDS, stays resident.
  for (int idx = tid; idx < COLS_WG * D_MODEL; idx += 256) {
    const int r = idx >> 10, k = idx & 1023;
    DT[r * DT_STR + k] = Dm[(size_t)(c0 + r) * D_MODEL + k];
  }

  for (int t = 0; t < SEQ; ++t) {
    // stage h (previous step's writes visible via grid barrier)
    for (int idx = tid; idx < 16 * D_MODEL; idx += 256) {
      const int r = idx >> 10, k = idx & 1023;
      HL[r * DT_STR + k] = HG[idx];
    }
    __syncthreads();

    // hB partials: wave w covers K in [128w, 128w+128)
    {
      v8f acc = {};
      const float* brow = Bmat + (size_t)l15 * D_MODEL;
      const float* hrow = HL + l15 * DT_STR;
      for (int k0 = w * 128; k0 < w * 128 + 128; k0 += 4) {
        const int kk = k0 + kh2;
        v2f av; av.x = hrow[kk]; av.y = hrow[kk + 1];
        v2f bv; bv.x = brow[kk]; bv.y = brow[kk + 1];
        acc = wmma_f32x4(av, bv, acc);
      }
#pragma unroll
      for (int i = 0; i < 8; ++i) SC[w * 256 + i * 32 + lane] = acc[i];
    }
    __syncthreads();
    // reduce 8 hB partials -> HB[m][r]
    {
      float s = 0.0f;
#pragma unroll
      for (int p = 0; p < 8; ++p) s += SC[p * 256 + tid];
      const int m = w + ((lane >> 4) << 3);
      HB[m * 20 + l15] = s;
    }
    __syncthreads();

    // acc partials: nt = w&1 (16 cols), kq = w>>1 (256 K each)
    {
      const int nt = w & 1, kq = w >> 1;
      const float* drow = DT + ((nt << 4) + l15) * DT_STR;
      const float* hrow = HL + l15 * DT_STR;
      v8f acc = {};
      for (int k0 = kq * 256; k0 < kq * 256 + 256; k0 += 4) {
        const int kk = k0 + kh2;
        v2f av; av.x = hrow[kk]; av.y = hrow[kk + 1];
        v2f bv; bv.x = drow[kk]; bv.y = drow[kk + 1];
        acc = wmma_f32x4(av, bv, acc);
      }
#pragma unroll
      for (int i = 0; i < 8; ++i) SC[w * 256 + i * 32 + lane] = acc[i];
    }
    __syncthreads();
    // reduce 4 K-quarters per N-tile -> SC tiles 8,9
    {
      const int nt = tid >> 7, e0 = tid & 127;
#pragma unroll
      for (int rep = 0; rep < 2; ++rep) {
        const int e = e0 + rep * 128;
        SC[(8 + nt) * 256 + e] = SC[nt * 256 + e] + SC[(nt + 2) * 256 + e] +
                                 SC[(nt + 4) * 256 + e] + SC[(nt + 6) * 256 + e];
      }
    }
    // low-rank: waves 0-1, lr = hB @ A^T[:, c0+16nt .. ) -> SC tiles 10,11
    if (w < 2) {
      const int nt = w;
      const float* arow = Amat + (size_t)(c0 + (nt << 4) + l15) * RANK;
      const float* hrow = HB + l15 * 20;
      v8f acc = {};
#pragma unroll
      for (int r0 = 0; r0 < RANK; r0 += 4) {
        const int kk = r0 + kh2;
        v2f av; av.x = hrow[kk]; av.y = hrow[kk + 1];
        v2f bv; bv.x = arow[kk]; bv.y = arow[kk + 1];
        acc = wmma_f32x4(av, bv, acc);
      }
#pragma unroll
      for (int i = 0; i < 8; ++i) SC[(10 + nt) * 256 + i * 32 + lane] = acc[i];
    }
    __syncthreads();

    // epilogue: thread (w, lane<16) owns (batch m = w, col n) for both tiles
    if (lane < 16) {
      const int m = w;
#pragma unroll
      for (int nt = 0; nt < 2; ++nt) {
        const int col = c0 + (nt << 4) + l15;
        const float av = SC[(8 + nt) * 256 + tid];
        const float lv = SC[(10 + nt) * 256 + tid];
        const size_t idx = ((size_t)m * SEQ + t) * D_MODEL + col;
        const float hn = gd * av + G[idx] * lv + Y[idx];   // Y holds u_t here
        Y[idx] = hn;                                       // becomes ys[t]
        HG[m * D_MODEL + col] = hn;
      }
    }
    grid_barrier(bar, N_WG);
  }
}

// ---------------------------------------------------------------------------
extern "C" void kernel_launch(void* const* d_in, const int* in_sizes, int n_in,
                              void* d_out, int out_size, void* d_ws, size_t ws_size,
                              hipStream_t stream) {
  (void)in_sizes; (void)n_in; (void)out_size; (void)ws_size;
  const float* x   = (const float*)d_in[0];
  const float* Sp  = (const float*)d_in[1];
  const float* gam = (const float*)d_in[2];
  const float* Am  = (const float*)d_in[3];
  const float* Bm  = (const float*)d_in[4];
  const float* gw  = (const float*)d_in[5];
  const float* gb  = (const float*)d_in[6];
  const float* ww  = (const float*)d_in[7];
  const float* wb  = (const float*)d_in[8];
  float* Y = (float*)d_out;

  // workspace layout (floats)
  float* W = (float*)d_ws;
  const size_t MM = (size_t)D_MODEL * D_MODEL;           // 1M floats
  float* Mb = W;            // I + S
  float* N0 = W + 1 * MM;   // I - S
  float* Xa = W + 2 * MM;   // Newton ping
  float* Xb = W + 3 * MM;   // Newton pong
  float* Tb = W + 4 * MM;   // M*X temp / residual
  float* Dm = W + 5 * MM;   // Cayley D
  float* HG = W + 6 * MM;   // h state, 16x1024 (rows 8..15 stay zero)
  unsigned* bar = (unsigned*)(W + 6 * MM + 16 * D_MODEL);
  float* G  = W + 7 * MM;   // gate values, 8*4096*1024
  __bf16* gwb = (__bf16*)(W + 39 * MM);    // bf16 weights (1M elems each)
  __bf16* wwb = gwb + MM;

  k_setup<<<4096, 256, 0, stream>>>(Sp, Mb, N0, Xa, HG, bar);
  k_cvt_w<<<4096, 256, 0, stream>>>(gw, ww, gwb, wwb);

  // Newton-Schulz: X <- X (2I - M X)
  float* Xc = Xa; float* Xn = Xb;
  for (int it = 0; it < NEWTON_ITERS; ++it) {
    k_gemm_nn<<<128, 256, 0, stream>>>(Tb, Mb, Xc);      // T = M*X
    k_neg2i<<<4096, 256, 0, stream>>>(Tb);               // T = 2I - T
    k_gemm_nn<<<128, 256, 0, stream>>>(Xn, Xc, Tb);      // X' = X*T
    float* tmp = Xc; Xc = Xn; Xn = tmp;
  }
  k_gemm_nn<<<128, 256, 0, stream>>>(Dm, Xc, N0);        // D = X * (I - S)

  // g -> G (ws), u -> Y (d_out, consumed in-place by the scan)
  k_proj<<<4096, 256, 0, stream>>>(x, gwb, gb, wwb, wb, G, Y);

  const size_t smem =
      (size_t)(COLS_WG * DT_STR + 16 * DT_STR + 12 * 256 + 16 * 20) * sizeof(float);
  k_scan<<<N_WG, 256, smem, stream>>>(Dm, Am, Bm, G, Y, HG, gam, bar);
}
